// KANLayer_65034394796222
// MI455X (gfx1250) — matrix operand order repaired and checked by reading, not compiled, not run on hardware
//
#include <hip/hip_runtime.h>

// ---------------------------------------------------------------------------
// KAN layer on gfx1250 as a one-hot GEMM via v_wmma_f32_16x16x32_f16.
//
//   out[r,i] = sum_j D[i, j, m(x[r,j])],  D[i,j,m] = C[i,j,m]+C[i,j,m+1]+C[i,j,m+2]
//   => out = A(one-hot, 16384 x 4608) * B(4608 x 512), k = m*512 + j, m in 0..8
//
// B split into f16 hi/lo (hi + lo ~ 22 mantissa bits), both accumulated into
// the same f32 WMMA accumulator -> near-f32 accuracy.
// Wave tile 32 rows x 64 cols: 2 A-frags x 4 B-frags -> 16 WMMA per K chunk.
// B fragments pipelined at distance TWO (each load completes ~8 WMMAs before
// use) + builtin prefetch two K chunks ahead to warm GL2.
// ---------------------------------------------------------------------------

typedef __attribute__((ext_vector_type(16))) _Float16 v16h;
typedef __attribute__((ext_vector_type(8)))  float    v8f;

#define IN_F    512
#define OUT_F   512
#define NSEG    9
#define NBASIS  12
#define KDIM    (NSEG * IN_F)        // 4608
#define KCHUNKS (KDIM / 32)          // 144
#define NTILES  (OUT_F / 16)         // 32
#define ROWS    16384                // 8 * 2048
#define MTILE   128                  // rows per workgroup
#define NTILE   256                  // cols per workgroup
#define PITCH   528                  // LDS row pitch (bytes), avoids bank conflicts

#define KC_STRIDE (32 * 32 * 16)     // halves per kc chunk image (16384)
#define NT_STRIDE (32 * 16)          // halves per nt within chunk (512)

// halves per B-fragment image: [kc][nt][lane][16]
#define BFRAG_HALVES ((size_t)KCHUNKS * NTILES * 32 * 16)   // 2,359,296

// numpy linspace(0,1,10) float32 knots: t[i] = float32(i * (1.0/9.0))
__device__ __forceinline__ int seg_of(float x) {
  constexpr double STEP = 1.0 / 9.0;
  constexpr float KN[8] = {
    (float)(1.0 * STEP), (float)(2.0 * STEP), (float)(3.0 * STEP),
    (float)(4.0 * STEP), (float)(5.0 * STEP), (float)(6.0 * STEP),
    (float)(7.0 * STEP), (float)(8.0 * STEP)
  };
  int m = 0;
#pragma unroll
  for (int i = 0; i < 8; ++i) m += (x >= KN[i]) ? 1 : 0;
  return m;   // 0..8 for x in [0,1)
}

// Build one-hot f16 A fragment (16x32 layout) from LDS segment codes.
__device__ __forceinline__ v16h build_a(const unsigned char* p, int m) {
  const unsigned long long q0 = *(const unsigned long long*)(p);
  const unsigned long long q1 = *(const unsigned long long*)(p + 16);
  v16h a;
#pragma unroll
  for (int h = 0; h < 8; ++h) {
    a[h]     = ((int)((q0 >> (8 * h)) & 0xFFull) == m) ? (_Float16)1.0f : (_Float16)0.0f;
    a[h + 8] = ((int)((q1 >> (8 * h)) & 0xFFull) == m) ? (_Float16)1.0f : (_Float16)0.0f;
  }
  return a;
}

// ---------------------------------------------------------------------------
// Kernel 1: build Bhi/Blo in WMMA B-fragment layout.
// B fragment (f16 32x16, wave32): VGPR v, lane l, half hf:
//   K = s + (s>=8 ? 8 : 0) + (l>=16 ? 8 : 0), s = 2*v+hf,  N = l & 15
// Stored as [kc][nt][lane][h] with h = 2*v+hf -> K = kc*32 + h + (h>=8 ? 8 : 0) + 8*laneHi.
// ---------------------------------------------------------------------------
__global__ __launch_bounds__(256) void kan_prep(const float* __restrict__ coeffs,
                                                _Float16* __restrict__ bhi,
                                                _Float16* __restrict__ blo) {
  const int gid  = blockIdx.x * 256 + threadIdx.x;   // 0 .. 147455
  const int lane = gid & 31;
  const int nt   = (gid >> 5) & 31;
  const int kc   = gid >> 10;                        // 0..143
  if (kc >= KCHUNKS) return;
  const int laneHi = lane >> 4;
  const int n = nt * 16 + (lane & 15);               // output feature i
  const size_t outBase = (size_t)gid * 16;
#pragma unroll
  for (int h = 0; h < 16; ++h) {
    const int K = kc * 32 + (h < 8 ? 8 * laneHi + h : 16 + 8 * laneHi + (h - 8));
    const int m = K >> 9;                            // segment 0..8
    const int j = K & 511;                           // input feature
    const float* c = coeffs + ((size_t)n * IN_F + j) * NBASIS + m;
    const float d = c[0] + c[1] + c[2];              // summed 3-hot coeffs
    const _Float16 hi = (_Float16)d;                 // RNE
    const _Float16 lo = (_Float16)(d - (float)hi);
    bhi[outBase + h] = hi;
    blo[outBase + h] = lo;
  }
}

// ---------------------------------------------------------------------------
// Kernel 2: one-hot GEMM. 512 threads = 16 waves, 4(M) x 4(N).
// Workgroup tile: 128 rows x 256 cols. Wave tile: 32 rows x 64 cols.
// ---------------------------------------------------------------------------
__global__ __launch_bounds__(512) void kan_gemm(const float* __restrict__ x,
                                                const _Float16* __restrict__ bhi,
                                                const _Float16* __restrict__ blo,
                                                float* __restrict__ out) {
  __shared__ __align__(16) unsigned char smem[MTILE * PITCH];

  const int tid = threadIdx.x;
  const size_t rowBase = (size_t)blockIdx.x * MTILE;

  // ---- Stage segment codes m(x) for this row tile into LDS (u8) ----
  for (int e = tid; e < MTILE * (IN_F / 4); e += 512) {
    const int row = e >> 7;          // 128 float4 per row
    const int c4  = e & 127;
    const float4 xv = ((const float4*)(x + (rowBase + row) * IN_F))[c4];
    const unsigned int pack = (unsigned)seg_of(xv.x)
                            | ((unsigned)seg_of(xv.y) << 8)
                            | ((unsigned)seg_of(xv.z) << 16)
                            | ((unsigned)seg_of(xv.w) << 24);
    *(unsigned int*)(smem + row * PITCH + c4 * 4) = pack;
  }
  __syncthreads();

  const int wave   = tid >> 5;
  const int lane   = tid & 31;
  const int wm     = wave & 3;       // M sub-tile (32 rows each)
  const int wn     = wave >> 2;      // N sub-tile (64 cols each)
  const int laneLo = lane & 15;
  const int laneHi = lane >> 4;

  // A fragment LDS offsets for the wave's two 16-row groups.
  const unsigned aOff0 = (unsigned)((wm * 32 + laneLo) * PITCH + (laneHi ? 8 : 0));
  const unsigned aOff1 = aOff0 + 16 * PITCH;

  // B fragment base: idx = ((kc*32 + nt)*32 + lane)*16, nt = ntBase + nf
  const int ntBase = blockIdx.y * 16 + wn * 4;
  const size_t bLaneBase = ((size_t)ntBase * 32 + lane) * 16;

  const v8f zero = {0.f, 0.f, 0.f, 0.f, 0.f, 0.f, 0.f, 0.f};
  v8f acc0[4], acc1[4];
#pragma unroll
  for (int nf = 0; nf < 4; ++nf) { acc0[nf] = zero; acc1[nf] = zero; }

  const _Float16* pbh = bhi + bLaneBase;
  const _Float16* pbl = blo + bLaneBase;

  // Distance-2 software pipeline: two fragment pairs in flight.
  v16h bhA = *(const v16h*)(pbh);                 // (kc=0, nf=0)
  v16h blA = *(const v16h*)(pbl);
  v16h bhB = *(const v16h*)(pbh + NT_STRIDE);     // (kc=0, nf=1)
  v16h blB = *(const v16h*)(pbl + NT_STRIDE);

  for (int kc = 0; kc < KCHUNKS; ++kc) {
    const int m  = kc >> 4;          // segment for this K chunk
    const int j0 = (kc & 15) * 32;   // input-feature base

    const v16h a0 = build_a(smem + aOff0 + j0, m);
    const v16h a1 = build_a(smem + aOff1 + j0, m);

    const _Float16* nbh = pbh + KC_STRIDE;   // next kc chunk
    const _Float16* nbl = pbl + KC_STRIDE;

#pragma unroll
    for (int nf = 0; nf < 4; ++nf) {
      const v16h cbh = bhA;
      const v16h cbl = blA;
      bhA = bhB;
      blA = blB;
      // Load the fragment TWO nf steps ahead (rolls into the next kc chunk).
      const _Float16* qh = (nf < 2) ? (pbh + (size_t)(nf + 2) * NT_STRIDE)
                                    : (nbh + (size_t)(nf - 2) * NT_STRIDE);
      const _Float16* ql = (nf < 2) ? (pbl + (size_t)(nf + 2) * NT_STRIDE)
                                    : (nbl + (size_t)(nf - 2) * NT_STRIDE);
      bhB = *(const v16h*)(qh);
      blB = *(const v16h*)(ql);
      // Prefetch two kc chunks ahead (warms GL2; schedules cleanly).
      __builtin_prefetch((const void*)(nbh + KC_STRIDE + (size_t)nf * NT_STRIDE), 0, 3);
      __builtin_prefetch((const void*)(nbl + KC_STRIDE + (size_t)nf * NT_STRIDE), 0, 3);

      acc0[nf] = __builtin_amdgcn_wmma_f32_16x16x32_f16(
          false, a0, false, cbh, (short)0, acc0[nf], false, false);
      acc1[nf] = __builtin_amdgcn_wmma_f32_16x16x32_f16(
          false, a1, false, cbh, (short)0, acc1[nf], false, false);
      acc0[nf] = __builtin_amdgcn_wmma_f32_16x16x32_f16(
          false, a0, false, cbl, (short)0, acc0[nf], false, false);
      acc1[nf] = __builtin_amdgcn_wmma_f32_16x16x32_f16(
          false, a1, false, cbl, (short)0, acc1[nf], false, false);
    }
    pbh = nbh;
    pbl = nbl;
  }

  // ---- Store: C/D f32 16x16 layout: M = v + 8*laneHi, N = laneLo ----
  const size_t rowW0   = rowBase + (size_t)wm * 32 + laneHi * 8;
  const size_t colBase = (size_t)blockIdx.y * 256 + wn * 64 + laneLo;
#pragma unroll
  for (int nf = 0; nf < 4; ++nf) {
    const size_t col = colBase + nf * 16;
#pragma unroll
    for (int v = 0; v < 8; ++v) {
      out[(rowW0 + v) * OUT_F + col]        = acc0[nf][v];
      out[(rowW0 + 16 + v) * OUT_F + col]   = acc1[nf][v];
    }
  }
}

extern "C" void kernel_launch(void* const* d_in, const int* in_sizes, int n_in,
                              void* d_out, int out_size, void* d_ws, size_t ws_size,
                              hipStream_t stream) {
  const float* x      = (const float*)d_in[0];   // (8,2048,512) f32
  const float* coeffs = (const float*)d_in[1];   // (512,512,12) f32
  float* out = (float*)d_out;                    // (8,2048,512) f32

  _Float16* bhi = (_Float16*)d_ws;               // 4.5 MB
  _Float16* blo = bhi + BFRAG_HALVES;            // 4.5 MB (+ pad slack beyond)

  // 144*32*32 = 147456 threads -> 576 blocks of 256
  kan_prep<<<dim3(576), dim3(256), 0, stream>>>(coeffs, bhi, blo);

  dim3 grid(ROWS / MTILE, OUT_F / NTILE);        // (128, 2)
  kan_gemm<<<grid, dim3(512), 0, stream>>>(x, bhi, blo, out);
}
